// Multi_Head_Attention_87299505258506
// MI455X (gfx1250) — compile-verified, matrix operand order checked by reading, not run on hardware
//
#include <hip/hip_runtime.h>
#include <hip/hip_bf16.h>

#define D_MODEL 1024
#define N_HEAD  16
#define HEAD_DIM 64
#define BS      4
#define SEQ     2048
#define MROWS   (BS * SEQ)          // 8192
#define NEG_INF (-1e9f)
#define LDP     72                  // padded LDS row stride in shorts (144B = 9*16B)

typedef __attribute__((ext_vector_type(16))) __bf16 v16bf;
typedef __attribute__((ext_vector_type(8)))  __bf16 v8bf;
typedef __attribute__((ext_vector_type(8)))  float  v8f;
typedef __attribute__((ext_vector_type(4)))  int    v4i;

// ---------- helpers ----------

__device__ __forceinline__ unsigned short f2bf(float f) {
    unsigned int u = __float_as_uint(f);
    unsigned int r = (u + 0x7FFFu + ((u >> 16) & 1u)) >> 16;   // RNE
    return (unsigned short)r;
}

__device__ __forceinline__ v8f zero8() {
    v8f z = {0.f, 0.f, 0.f, 0.f, 0.f, 0.f, 0.f, 0.f};
    return z;
}

// Load one 16x32 bf16 A-fragment (or B-fragment of a K-contiguous matrix).
// Layout (ISA 7.12.2, 16-bit A 16x32): lane<16 -> K = k0+0..7 and k0+16..23,
// lane>=16 -> K = k0+8..15 and k0+24..31. "row" is the full row/col index.
__device__ __forceinline__ v16bf load_frag(const unsigned short* base,
                                           int stride, int row, int k0, int lane) {
    int kb = k0 + ((lane & 16) ? 8 : 0);
    const __bf16* p = (const __bf16*)(base + (size_t)row * stride + kb);
    v8bf lo = *(const v8bf*)p;
    v8bf hi = *(const v8bf*)(p + 16);
    return __builtin_shufflevector(lo, hi, 0, 1, 2, 3, 4, 5, 6, 7,
                                           8, 9, 10, 11, 12, 13, 14, 15);
}

__device__ __forceinline__ v8f wmma_bf16(v16bf a, v16bf b, v8f c) {
    return __builtin_amdgcn_wmma_f32_16x16x32_bf16(
        false, a, false, b, (short)0, c, false, false);
}

__device__ __forceinline__ float rmax16(float v) {
    v = fmaxf(v, __shfl_xor(v, 1, 32));
    v = fmaxf(v, __shfl_xor(v, 2, 32));
    v = fmaxf(v, __shfl_xor(v, 4, 32));
    v = fmaxf(v, __shfl_xor(v, 8, 32));
    return v;
}
__device__ __forceinline__ float rsum16(float v) {
    v += __shfl_xor(v, 1, 32);
    v += __shfl_xor(v, 2, 32);
    v += __shfl_xor(v, 4, 32);
    v += __shfl_xor(v, 8, 32);
    return v;
}

// ---------- gfx1250 async copy: global -> LDS, 16 bytes per lane ----------

typedef __attribute__((address_space(1))) v4i as1_v4i;   // global
typedef __attribute__((address_space(3))) v4i as3_v4i;   // LDS

#if __has_builtin(__builtin_amdgcn_global_load_async_to_lds_b128)
#define HAVE_ASYNC_LDS 1
#endif

__device__ __forceinline__ void async_cp16(const unsigned short* g, unsigned short* l) {
#ifdef HAVE_ASYNC_LDS
    __builtin_amdgcn_global_load_async_to_lds_b128((as1_v4i*)g, (as3_v4i*)l, 0, 0);
#else
    *(v8bf*)l = *(const v8bf*)g;   // fallback: sync load + ds_write
#endif
}

#define ASYNC_WAIT_4 asm volatile("s_wait_asynccnt 0x4" ::: "memory")
#define ASYNC_WAIT_0 asm volatile("s_wait_asynccnt 0x0" ::: "memory")
#define DS_WAIT_0    asm volatile("s_wait_dscnt 0x0"    ::: "memory")

// ---------- kernel 1: f32 -> bf16 convert ----------

__global__ void f32_to_bf16_kernel(const float* __restrict__ src,
                                   unsigned short* __restrict__ dst, int n) {
    int i = blockIdx.x * blockDim.x + threadIdx.x;
    int stride = gridDim.x * blockDim.x;
    for (; i < n; i += stride) dst[i] = f2bf(src[i]);
}

// ---------- kernel 2: GEMM  Y = A @ W^T + bias ----------
// A: MROWS x 1024 bf16. W: 1024 x 1024 bf16 (output col n contracts with W
// row n, contiguous -> clean B-frag). Waves in a block share n0 (weight reuse).
// mode 0: Q -> bf16 [b,h,s,64], scaled by 0.125
// mode 1: K -> bf16 [b,h,s,64]
// mode 2: V -> bf16 [b,h,64,s]  (transposed per head)
// mode 3: f32 [m,n] to out32 (final projection)
__global__ __launch_bounds__(256) void gemm_bf16_kernel(
    const unsigned short* __restrict__ A,
    const unsigned short* __restrict__ W,
    const float* __restrict__ bias,
    unsigned short* __restrict__ out16,
    float* __restrict__ out32,
    int mode) {
    const int lane = threadIdx.x & 31;
    const int wave = threadIdx.x >> 5;
    const int tn   = blockIdx.x >> 5;                    // 16 N-tiles of 64
    const int tm   = ((blockIdx.x & 31) << 3) + wave;    // 256 M-tiles of 32
    const int m0   = tm * 32;
    const int n0   = tn * 64;
    const int c    = lane & 15;
    const int half = lane >> 4;

    v8f acc[2][4];
#pragma unroll
    for (int i = 0; i < 2; ++i)
#pragma unroll
        for (int j = 0; j < 4; ++j) acc[i][j] = zero8();

    for (int k0 = 0; k0 < D_MODEL; k0 += 32) {
        v16bf a0 = load_frag(A, D_MODEL, m0 + c,      k0, lane);
        v16bf a1 = load_frag(A, D_MODEL, m0 + 16 + c, k0, lane);
        v16bf b0 = load_frag(W, D_MODEL, n0 + 0  + c, k0, lane);
        v16bf b1 = load_frag(W, D_MODEL, n0 + 16 + c, k0, lane);
        v16bf b2 = load_frag(W, D_MODEL, n0 + 32 + c, k0, lane);
        v16bf b3 = load_frag(W, D_MODEL, n0 + 48 + c, k0, lane);
        acc[0][0] = wmma_bf16(a0, b0, acc[0][0]);
        acc[0][1] = wmma_bf16(a0, b1, acc[0][1]);
        acc[0][2] = wmma_bf16(a0, b2, acc[0][2]);
        acc[0][3] = wmma_bf16(a0, b3, acc[0][3]);
        acc[1][0] = wmma_bf16(a1, b0, acc[1][0]);
        acc[1][1] = wmma_bf16(a1, b1, acc[1][1]);
        acc[1][2] = wmma_bf16(a1, b2, acc[1][2]);
        acc[1][3] = wmma_bf16(a1, b3, acc[1][3]);
    }

    const float scale = (mode == 0) ? 0.125f : 1.0f;   // 1/sqrt(HEAD_DIM)
#pragma unroll
    for (int mt = 0; mt < 2; ++mt) {
#pragma unroll
        for (int nt = 0; nt < 4; ++nt) {
            const int ncol = n0 + nt * 16 + c;
            const float bn = bias[ncol];
#pragma unroll
            for (int j = 0; j < 8; ++j) {
                const int mrow = m0 + mt * 16 + j + half * 8;
                float v = (acc[mt][nt][j] + bn) * scale;
                if (mode <= 1) {            // [b,h,s,dh]
                    int bb = mrow >> 11, s = mrow & 2047;
                    int h = ncol >> 6, dh = ncol & 63;
                    out16[(((size_t)(bb * N_HEAD + h) * SEQ + s) << 6) + dh] = f2bf(v);
                } else if (mode == 2) {     // [b,h,dh,s]
                    int bb = mrow >> 11, s = mrow & 2047;
                    int h = ncol >> 6, dh = ncol & 63;
                    out16[(((size_t)(bb * N_HEAD + h) * HEAD_DIM + dh) << 11) + s] = f2bf(v);
                } else {                    // f32 [m,n]
                    out32[(size_t)mrow * D_MODEL + ncol] = v;
                }
            }
        }
    }
}

// ---------- kernel 3: flash attention ----------
// Block = 8 waves = 128 queries of one (b,h). K/V 64-key blocks are staged
// into LDS once per block via async global->LDS copies, double-buffered so
// the next block's DMA overlaps this block's WMMAs + softmax.

__global__ __launch_bounds__(256) void attn_kernel(
    const unsigned short* __restrict__ Q,   // [B,H,S,64] bf16, pre-scaled
    const unsigned short* __restrict__ Km,  // [B,H,S,64] bf16
    const unsigned short* __restrict__ Vt,  // [B,H,64,S] bf16
    const int* __restrict__ kpm,            // [B,S]
    unsigned short* __restrict__ O)         // [B,S,1024] bf16 (heads merged)
{
    __shared__ __align__(16) unsigned short Kbuf[2][64 * LDP];   // 2 x 9 KB
    __shared__ __align__(16) unsigned short Vbuf[2][64 * LDP];   // 2 x 9 KB
    __shared__ __align__(16) unsigned short Pbuf[8][16 * LDP];   // 18 KB

    const int tid  = threadIdx.x;
    const int lane = tid & 31;
    const int wave = tid >> 5;
    const int c    = lane & 15;
    const int half = lane >> 4;
    const int h    = blockIdx.y;
    const int b    = blockIdx.z;
    const int q0   = blockIdx.x * 128 + wave * 16;

    const unsigned short* Qbh = Q  + (size_t)(b * N_HEAD + h) * SEQ * HEAD_DIM;
    const unsigned short* Kbh = Km + (size_t)(b * N_HEAD + h) * SEQ * HEAD_DIM;
    const unsigned short* Vbh = Vt + (size_t)(b * N_HEAD + h) * HEAD_DIM * SEQ;
    const int* mbase = kpm + b * SEQ;
    unsigned short* myP = &Pbuf[wave][0];

    // cooperative stage of one 64-key K block + V block (4 async b128/thread)
    auto stage = [&](int bi, int kb0) {
#pragma unroll
        for (int i = 0; i < 2; ++i) {
            const int q  = tid + i * 256;        // 512 16B-chunks per matrix
            const int r  = q >> 3;               // K: key row / V: head-dim row
            const int ch = (q & 7) * 8;
            async_cp16(Kbh + (size_t)(kb0 + r) * HEAD_DIM + ch, &Kbuf[bi][r * LDP + ch]);
            async_cp16(Vbh + (size_t)r * SEQ + kb0 + ch,        &Vbuf[bi][r * LDP + ch]);
        }
    };

    const v16bf qf0 = load_frag(Qbh, HEAD_DIM, q0 + c, 0,  lane);
    const v16bf qf1 = load_frag(Qbh, HEAD_DIM, q0 + c, 32, lane);

    v8f o[4];
#pragma unroll
    for (int t = 0; t < 4; ++t) o[t] = zero8();
    float mrow[8], lrow[8];
#pragma unroll
    for (int j = 0; j < 8; ++j) { mrow[j] = -1e30f; lrow[j] = 0.f; }

    stage(0, 0);
    for (int it = 0; it < SEQ / 64; ++it) {
        const int kb0 = it * 64;
        const unsigned short* Kl = &Kbuf[it & 1][0];
        const unsigned short* Vl = &Vbuf[it & 1][0];
        if (it + 1 < SEQ / 64) {
            stage((it + 1) & 1, kb0 + 64);   // prefetch next block
            ASYNC_WAIT_4;                    // current block's 4 copies done
        } else {
            ASYNC_WAIT_0;
        }
        __syncthreads();

        // S = (Q/sqrt(d)) @ K^T : 16 queries x 64 keys, masked
        v8f s[4];
#pragma unroll
        for (int t = 0; t < 4; ++t) {
            v16bf kf0 = load_frag(Kl, LDP, t * 16 + c, 0,  lane);
            v16bf kf1 = load_frag(Kl, LDP, t * 16 + c, 32, lane);
            v8f st = zero8();
            st = wmma_bf16(qf0, kf0, st);
            st = wmma_bf16(qf1, kf1, st);
            const int mk = mbase[kb0 + t * 16 + c];   // column mask (per lane)
#pragma unroll
            for (int j = 0; j < 8; ++j) st[j] = mk ? NEG_INF : st[j];
            s[t] = st;
        }
        // online softmax (rows live across 16-lane halves)
        float mnew[8], alpha[8];
#pragma unroll
        for (int j = 0; j < 8; ++j) {
            float v = fmaxf(fmaxf(s[0][j], s[1][j]), fmaxf(s[2][j], s[3][j]));
            v = rmax16(v);
            mnew[j]  = fmaxf(mrow[j], v);
            alpha[j] = __expf(mrow[j] - mnew[j]);
            mrow[j]  = mnew[j];
        }
#pragma unroll
        for (int j = 0; j < 8; ++j) {
            float rs = 0.f;
#pragma unroll
            for (int t = 0; t < 4; ++t) {
                float p = __expf(s[t][j] - mnew[j]);
                s[t][j] = p;
                rs += p;
            }
            rs = rsum16(rs);
            lrow[j] = lrow[j] * alpha[j] + rs;
        }
#pragma unroll
        for (int t = 0; t < 4; ++t)
#pragma unroll
            for (int j = 0; j < 8; ++j) o[t][j] *= alpha[j];

        // C-layout P -> A-layout via wave-private LDS (bf16)
#pragma unroll
        for (int t = 0; t < 4; ++t)
#pragma unroll
            for (int j = 0; j < 8; ++j)
                myP[(j + half * 8) * LDP + t * 16 + c] = f2bf(s[t][j]);
        DS_WAIT_0;

        v16bf pf0 = load_frag(myP, LDP, c, 0,  lane);
        v16bf pf1 = load_frag(myP, LDP, c, 32, lane);

        // O += P @ V  (LDS Vt rows contiguous over keys)
#pragma unroll
        for (int dt = 0; dt < 4; ++dt) {
            v16bf vf0 = load_frag(Vl, LDP, dt * 16 + c, 0,  lane);
            v16bf vf1 = load_frag(Vl, LDP, dt * 16 + c, 32, lane);
            o[dt] = wmma_bf16(pf0, vf0, o[dt]);
            o[dt] = wmma_bf16(pf1, vf1, o[dt]);
        }
        __syncthreads();   // all reads done before next stage overwrites
    }

    float inv[8];
#pragma unroll
    for (int j = 0; j < 8; ++j) inv[j] = 1.0f / lrow[j];
#pragma unroll
    for (int t = 0; t < 4; ++t)
#pragma unroll
        for (int j = 0; j < 8; ++j) {
            const int qrow = q0 + j + half * 8;
            const int d    = h * HEAD_DIM + t * 16 + c;
            O[(size_t)(b * SEQ + qrow) * D_MODEL + d] = f2bf(o[t][j] * inv[j]);
        }
}

// ---------- launch ----------

extern "C" void kernel_launch(void* const* d_in, const int* in_sizes, int n_in,
                              void* d_out, int out_size, void* d_ws, size_t ws_size,
                              hipStream_t stream) {
    const float* X   = (const float*)d_in[0];
    const int*   kpm = (const int*)d_in[1];
    const float* WQ  = (const float*)d_in[2];
    const float* bQ  = (const float*)d_in[3];
    const float* WK  = (const float*)d_in[4];
    const float* bK  = (const float*)d_in[5];
    const float* WV  = (const float*)d_in[6];
    const float* bV  = (const float*)d_in[7];
    const float* WO  = (const float*)d_in[8];
    const float* bO  = (const float*)d_in[9];

    char* ws = (char*)d_ws;
    const size_t XB  = (size_t)MROWS * D_MODEL * 2;     // 16 MB
    const size_t WB  = (size_t)D_MODEL * D_MODEL * 2;   // 2 MB
    unsigned short* Xb  = (unsigned short*)(ws);
    unsigned short* Wqb = (unsigned short*)(ws + XB);
    unsigned short* Wkb = (unsigned short*)(ws + XB + WB);
    unsigned short* Wvb = (unsigned short*)(ws + XB + 2 * WB);
    unsigned short* Wob = (unsigned short*)(ws + XB + 3 * WB);
    unsigned short* Qb  = (unsigned short*)(ws + XB + 4 * WB);
    unsigned short* Kb  = (unsigned short*)(ws + 2 * XB + 4 * WB);
    unsigned short* Vtb = (unsigned short*)(ws + 3 * XB + 4 * WB);
    unsigned short* Ab  = (unsigned short*)(ws + 4 * XB + 4 * WB);

    f32_to_bf16_kernel<<<4096, 256, 0, stream>>>(X,  Xb,  MROWS * D_MODEL);
    f32_to_bf16_kernel<<<1024, 256, 0, stream>>>(WQ, Wqb, D_MODEL * D_MODEL);
    f32_to_bf16_kernel<<<1024, 256, 0, stream>>>(WK, Wkb, D_MODEL * D_MODEL);
    f32_to_bf16_kernel<<<1024, 256, 0, stream>>>(WV, Wvb, D_MODEL * D_MODEL);
    f32_to_bf16_kernel<<<1024, 256, 0, stream>>>(WO, Wob, D_MODEL * D_MODEL);

    // (MROWS/32)*(D_MODEL/64) = 4096 waves / 8 waves per block
    gemm_bf16_kernel<<<512, 256, 0, stream>>>(Xb, Wqb, bQ, Qb,  nullptr, 0);
    gemm_bf16_kernel<<<512, 256, 0, stream>>>(Xb, Wkb, bK, Kb,  nullptr, 1);
    gemm_bf16_kernel<<<512, 256, 0, stream>>>(Xb, Wvb, bV, Vtb, nullptr, 2);

    attn_kernel<<<dim3(SEQ / 128, N_HEAD, BS), 256, 0, stream>>>(Qb, Kb, Vtb, kpm, Ab);

    gemm_bf16_kernel<<<512, 256, 0, stream>>>(Ab, Wob, bO, nullptr, (float*)d_out, 3);
}
